// MultiHeadAttn_82712480186713
// MI455X (gfx1250) — matrix operand order repaired and checked by reading, not compiled, not run on hardware
//
#include <hip/hip_runtime.h>

#define S_LEN  2048
#define DMODEL 1024
#define NHEAD  16
#define DHEAD  64
#define NBATCH 2
#define PAVG_LD 2052  // 2048 + 4 pad: rows m and m+8 hit disjoint LDS banks

typedef __attribute__((ext_vector_type(16))) __bf16 v16bf;
typedef __attribute__((ext_vector_type(8)))  __bf16 v8bf;
typedef __attribute__((ext_vector_type(8)))  float  v8f;

__device__ __forceinline__ __bf16 to_bf16(float x) { return (__bf16)x; }

__device__ __forceinline__ v8f wmma_bf16(v16bf a, v16bf b, v8f c) {
  // v_wmma_f32_16x16x32_bf16: (neg_a, A, neg_b, B, c_mod, C, reuse_a, reuse_b)
  return __builtin_amdgcn_wmma_f32_16x16x32_bf16(false, a, false, b, (short)0, c,
                                                 false, false);
}

// A fragment: 16(M) x 32(K) bf16 from a row-major [M][K] source (ld elements).
// Per ISA layout: lanes 0-15 hold M=lane; group g=lane>>4 holds K chunks
// [8g .. 8g+7] in VGPR0..3 and [16+8g .. 23+8g] in VGPR4..7.
__device__ __forceinline__ v16bf load_frag_a(const __bf16* base, int ld) {
  const int lane = threadIdx.x & 31;
  const int r = lane & 15, g = lane >> 4;
  const __bf16* p = base + (size_t)r * ld + 8 * g;
  v8bf c0 = *(const v8bf*)(p);
  v8bf c1 = *(const v8bf*)(p + 16);
  v16bf f;
#pragma unroll
  for (int i = 0; i < 8; ++i) { f[i] = c0[i]; f[i + 8] = c1[i]; }
  return f;
}

// B fragment: 32(K) x 16(N) bf16 loaded from a TRANSPOSED [N][K] source.
// Per ISA layout: lanes hold N=lane&15; group g=lane>>4 holds K chunks
// [8g .. 8g+7] in VGPR0..3 and [16+8g .. 23+8g] in VGPR4..7.
__device__ __forceinline__ v16bf load_frag_b(const __bf16* baseT, int ld) {
  const int lane = threadIdx.x & 31;
  const int n = lane & 15, g = lane >> 4;
  const __bf16* p = baseT + (size_t)n * ld + 8 * g;
  v8bf c0 = *(const v8bf*)(p);
  v8bf c1 = *(const v8bf*)(p + 16);
  v16bf f;
#pragma unroll
  for (int i = 0; i < 8; ++i) { f[i] = c0[i]; f[i + 8] = c1[i]; }
  return f;
}

// ---------------------------------------------------------------------------
// Elementwise prep kernels
// ---------------------------------------------------------------------------
__global__ void cast_bf16_kernel(const float* __restrict__ in,
                                 __bf16* __restrict__ out, size_t n) {
  for (size_t i = (size_t)blockIdx.x * blockDim.x + threadIdx.x; i < n;
       i += (size_t)gridDim.x * blockDim.x)
    out[i] = to_bf16(in[i]);
}

// Wt[n*K + k] = bf16(W[k*N + n])   (W is [K][N] row-major)
__global__ void transpose_cast_kernel(const float* __restrict__ W,
                                      __bf16* __restrict__ Wt, int K, int N) {
  const size_t total = (size_t)K * N;
  for (size_t i = (size_t)blockIdx.x * blockDim.x + threadIdx.x; i < total;
       i += (size_t)gridDim.x * blockDim.x) {
    size_t n = i / K, k = i % K;
    Wt[i] = to_bf16(W[k * (size_t)N + n]);
  }
}

// ---------------------------------------------------------------------------
// Generic bf16 WMMA GEMM: C = A[M,K] @ Bt^T + bias, Bt stored [N][K] bf16.
// 256 threads = 8 waves; block tile 128x128; wave tile 32x64 (2x4 frags).
// L2-resident operands -> stream fragments directly from global (b128 loads).
// mode 0: bf16 out as [B,H,S,dh]    (Q/K projection, scale folds 1/sqrt(dh))
// mode 1: bf16 out as [B,H,dh,S]    (V projection, key-contiguous for PV B-frags)
// mode 3: f32 out row-major [M,N]   (final FC -> d_out)
// ---------------------------------------------------------------------------
__global__ __launch_bounds__(256) void gemm_bf16_kernel(
    const __bf16* __restrict__ A, const __bf16* __restrict__ Bt,
    const float* __restrict__ bias, void* __restrict__ outp,
    int M, int N, int K, int mode, float scale) {
  const int wave = threadIdx.x >> 5;
  const int row0 = blockIdx.x * 128 + (wave >> 1) * 32;
  const int col0 = blockIdx.y * 128 + (wave & 1) * 64;

  v8f acc[2][4];
#pragma unroll
  for (int i = 0; i < 2; ++i)
#pragma unroll
    for (int j = 0; j < 4; ++j)
#pragma unroll
      for (int e = 0; e < 8; ++e) acc[i][j][e] = 0.0f;

  for (int kk = 0; kk < K; kk += 32) {
    v16bf af0 = load_frag_a(A + (size_t)row0 * K + kk, K);
    v16bf af1 = load_frag_a(A + (size_t)(row0 + 16) * K + kk, K);
    v16bf bf0 = load_frag_b(Bt + (size_t)(col0 +  0) * K + kk, K);
    v16bf bf1 = load_frag_b(Bt + (size_t)(col0 + 16) * K + kk, K);
    v16bf bf2 = load_frag_b(Bt + (size_t)(col0 + 32) * K + kk, K);
    v16bf bf3 = load_frag_b(Bt + (size_t)(col0 + 48) * K + kk, K);
    acc[0][0] = wmma_bf16(af0, bf0, acc[0][0]);
    acc[0][1] = wmma_bf16(af0, bf1, acc[0][1]);
    acc[0][2] = wmma_bf16(af0, bf2, acc[0][2]);
    acc[0][3] = wmma_bf16(af0, bf3, acc[0][3]);
    acc[1][0] = wmma_bf16(af1, bf0, acc[1][0]);
    acc[1][1] = wmma_bf16(af1, bf1, acc[1][1]);
    acc[1][2] = wmma_bf16(af1, bf2, acc[1][2]);
    acc[1][3] = wmma_bf16(af1, bf3, acc[1][3]);
  }

  // C/D layout: n = lane&15, m = vgpr + 8*(lane>>4)
  const int lane = threadIdx.x & 31;
  const int nn = lane & 15, hi = lane >> 4;
#pragma unroll
  for (int i = 0; i < 2; ++i)
#pragma unroll
    for (int j = 0; j < 4; ++j)
#pragma unroll
      for (int r = 0; r < 8; ++r) {
        const int m = row0 + i * 16 + r + 8 * hi;
        const int c = col0 + j * 16 + nn;
        const float v = (acc[i][j][r] + bias[c]) * scale;
        if (mode == 0) {
          const int b = m >> 11, s = m & (S_LEN - 1);
          const int h = c >> 6, d = c & (DHEAD - 1);
          ((__bf16*)outp)[((size_t)(b * NHEAD + h) * S_LEN + s) * DHEAD + d] =
              to_bf16(v);
        } else if (mode == 1) {
          const int b = m >> 11, s = m & (S_LEN - 1);
          const int h = c >> 6, d = c & (DHEAD - 1);
          ((__bf16*)outp)[((size_t)(b * NHEAD + h) * DHEAD + d) * S_LEN + s] =
              to_bf16(v);
        } else {
          ((float*)outp)[(size_t)m * N + c] = v;
        }
      }
}

// ---------------------------------------------------------------------------
// Attention: one workgroup (16 waves / 512 threads) per (batch, 16-row q tile);
// wave w handles head w. Two-pass streaming softmax per wave; the head-average
// of probabilities is reduced in a 128 KB LDS tile with ds_add_f32 atomics
// (CDNA5: 320 KB LDS per workgroup), then written out once, coalesced.
// ---------------------------------------------------------------------------
__global__ __launch_bounds__(512) void attn_kernel(
    const __bf16* __restrict__ Qp,   // [B,H,S,dh] bf16, pre-scaled by 0.125
    const __bf16* __restrict__ Kp,   // [B,H,S,dh] bf16
    const __bf16* __restrict__ Vt,   // [B,H,dh,S] bf16
    const int* __restrict__ pmask,   // [B,S]
    float* __restrict__ attn_avg,    // [B,S,S] f32
    __bf16* __restrict__ Obf) {      // [B*S, D] bf16
  const int qt = blockIdx.x, b = blockIdx.y;
  const int m0 = qt * 16;
  const int tid = threadIdx.x;
  const int h = tid >> 5;  // one wave per head
  const int lane = tid & 31;
  const int nn = lane & 15, hi = lane >> 4;

  __shared__ float  pavg[16][PAVG_LD];     // 131,328 B head-avg accumulator
  __shared__ __bf16 pbf[NHEAD][16 * 32];   //  16,384 B per-wave P staging

  // cooperative zero of the shared accumulator (incl. padding)
  for (int i = tid; i < 16 * PAVG_LD; i += 512) ((float*)pavg)[i] = 0.0f;
  __syncthreads();

  const size_t head = (size_t)(b * NHEAD + h);
  const __bf16* Qbase = Qp + (head * S_LEN + m0) * DHEAD;
  const __bf16* Kbase = Kp + head * S_LEN * DHEAD;
  const __bf16* Vbase = Vt + head * DHEAD * S_LEN;
  const int* mrow = pmask + (size_t)b * S_LEN;

  const v16bf qf0 = load_frag_a(Qbase, DHEAD);
  const v16bf qf1 = load_frag_a(Qbase + 32, DHEAD);

  float mx[8], ls[8];
#pragma unroll
  for (int r = 0; r < 8; ++r) { mx[r] = -1.0e30f; ls[r] = 0.0f; }

  // ---- pass 1: per-lane online max / exp-sum over this lane's columns ----
  for (int kt = 0; kt < S_LEN / 16; ++kt) {
    v16bf kf0 = load_frag_b(Kbase + (size_t)(kt * 16) * DHEAD, DHEAD);
    v16bf kf1 = load_frag_b(Kbase + (size_t)(kt * 16) * DHEAD + 32, DHEAD);
    v8f st;
#pragma unroll
    for (int e = 0; e < 8; ++e) st[e] = 0.0f;
    st = wmma_bf16(qf0, kf0, st);
    st = wmma_bf16(qf1, kf1, st);
    const bool valid = (mrow[kt * 16 + nn] != 0);
#pragma unroll
    for (int r = 0; r < 8; ++r) {
      const float v = valid ? st[r] : -1.0e30f;
      const float mo = mx[r];
      const float mn = fmaxf(mo, v);
      ls[r] = ls[r] * __expf(mo - mn) + __expf(v - mn);
      mx[r] = mn;
    }
  }
  // combine across the 16 lanes that share a row (xor 1,2,4,8 stays in-group)
#pragma unroll
  for (int r = 0; r < 8; ++r) {
#pragma unroll
    for (int d = 1; d < 16; d <<= 1) {
      const float om = __shfl_xor(mx[r], d);
      const float ol = __shfl_xor(ls[r], d);
      const float mn = fmaxf(mx[r], om);
      ls[r] = ls[r] * __expf(mx[r] - mn) + ol * __expf(om - mn);
      mx[r] = mn;
    }
  }
  float inv[8];
#pragma unroll
  for (int r = 0; r < 8; ++r) inv[r] = 1.0f / ls[r];

  // ---- pass 2: P tiles -> LDS head-average (ds_add_f32), P(bf16) @ V ----
  v8f oacc[4];
#pragma unroll
  for (int j = 0; j < 4; ++j)
#pragma unroll
    for (int e = 0; e < 8; ++e) oacc[j][e] = 0.0f;

  for (int kt2 = 0; kt2 < S_LEN / 32; ++kt2) {
#pragma unroll
    for (int sub = 0; sub < 2; ++sub) {
      const int kt = kt2 * 2 + sub;
      v16bf kf0 = load_frag_b(Kbase + (size_t)(kt * 16) * DHEAD, DHEAD);
      v16bf kf1 = load_frag_b(Kbase + (size_t)(kt * 16) * DHEAD + 32, DHEAD);
      v8f st;
#pragma unroll
      for (int e = 0; e < 8; ++e) st[e] = 0.0f;
      st = wmma_bf16(qf0, kf0, st);
      st = wmma_bf16(qf1, kf1, st);
      const int ng = kt * 16 + nn;
      const bool valid = (mrow[ng] != 0);
#pragma unroll
      for (int r = 0; r < 8; ++r) {
        const float p = valid ? __expf(st[r] - mx[r]) * inv[r] : 0.0f;
        const int m = r + 8 * hi;
        atomicAdd(&pavg[m][ng], p * (1.0f / NHEAD));  // conflict-free ds_add
        pbf[h][m * 32 + sub * 16 + nn] = to_bf16(p);
      }
    }
    // per-wave staging region: DS ops from one wave execute in order at the
    // LDS, so the A-fragment reload below needs no barrier.
    const v16bf pa = load_frag_a(pbf[h], 32);  // P as 16x32 A fragment
#pragma unroll
    for (int j = 0; j < 4; ++j) {
      v16bf vf = load_frag_b(Vbase + (size_t)(j * 16) * S_LEN + kt2 * 32, S_LEN);
      oacc[j] = wmma_bf16(pa, vf, oacc[j]);
    }
  }

  // ---- single coalesced writeout of the head-averaged attention tile ----
  __syncthreads();
  float* dst = attn_avg + ((size_t)b * S_LEN + m0) * S_LEN;
  for (int i = tid; i < 16 * (S_LEN / 4); i += 512) {
    const int m = i >> 9;            // 512 float4's per row
    const int nq = i & 511;
    const float4 vq = *(const float4*)&pavg[m][nq * 4];
    *(float4*)&dst[(size_t)m * S_LEN + nq * 4] = vq;
  }

  // store O tile (token-major bf16, feeds the FC GEMM)
#pragma unroll
  for (int j = 0; j < 4; ++j)
#pragma unroll
    for (int r = 0; r < 8; ++r) {
      const int m = m0 + r + 8 * hi;
      const int c = h * DHEAD + j * 16 + nn;
      Obf[((size_t)b * S_LEN + m) * DMODEL + c] = to_bf16(oacc[j][r]);
    }
}

// ---------------------------------------------------------------------------
extern "C" void kernel_launch(void* const* d_in, const int* in_sizes, int n_in,
                              void* d_out, int out_size, void* d_ws,
                              size_t ws_size, hipStream_t stream) {
  (void)in_sizes; (void)n_in; (void)out_size; (void)ws_size;
  const float* q   = (const float*)d_in[0];
  const float* k   = (const float*)d_in[1];
  const float* v   = (const float*)d_in[2];
  const float* Wq  = (const float*)d_in[3];
  const float* bq  = (const float*)d_in[4];
  const float* Wk  = (const float*)d_in[5];
  const float* bk  = (const float*)d_in[6];
  const float* Wv  = (const float*)d_in[7];
  const float* bv  = (const float*)d_in[8];
  const float* Wfc = (const float*)d_in[9];
  const float* bfc = (const float*)d_in[10];
  const int* pmask = (const int*)d_in[11];

  float* outp = (float*)d_out;                                  // [B,S,2D]
  float* attn_avg = outp + (size_t)NBATCH * S_LEN * 2 * DMODEL; // [B,S,S]

  const size_t MTOK = (size_t)NBATCH * S_LEN;  // 4096

  char* ws = (char*)d_ws;
  size_t off = 0;
  auto bump = [&](size_t elems) -> __bf16* {
    __bf16* p = (__bf16*)(ws + off);
    off += (elems * sizeof(__bf16) + 255) & ~(size_t)255;
    return p;
  };
  __bf16* qbf  = bump(MTOK * DMODEL);
  __bf16* kbf  = bump(MTOK * DMODEL);
  __bf16* vbf  = bump(MTOK * 2 * DMODEL);
  __bf16* WqT  = bump((size_t)DMODEL * DMODEL);
  __bf16* WkT  = bump((size_t)DMODEL * DMODEL);
  __bf16* WvT  = bump((size_t)DMODEL * 2 * DMODEL);   // [N=1024][K=2048]
  __bf16* WfcT = bump((size_t)2 * DMODEL * DMODEL);   // [N=2048][K=1024]
  __bf16* Qp   = bump(MTOK * DMODEL);                 // [B,H,S,dh]
  __bf16* Kp   = bump(MTOK * DMODEL);                 // [B,H,S,dh]
  __bf16* Vt   = bump(MTOK * DMODEL);                 // [B,H,dh,S]
  __bf16* Obf  = bump(MTOK * DMODEL);                 // [B*S, D]

  // bf16 casts of activations
  cast_bf16_kernel<<<2048, 256, 0, stream>>>(q, qbf, MTOK * DMODEL);
  cast_bf16_kernel<<<2048, 256, 0, stream>>>(k, kbf, MTOK * DMODEL);
  cast_bf16_kernel<<<4096, 256, 0, stream>>>(v, vbf, MTOK * 2 * DMODEL);
  // weight transposes to [N][K] bf16 for B-fragment streaming
  transpose_cast_kernel<<<2048, 256, 0, stream>>>(Wq, WqT, DMODEL, DMODEL);
  transpose_cast_kernel<<<2048, 256, 0, stream>>>(Wk, WkT, DMODEL, DMODEL);
  transpose_cast_kernel<<<4096, 256, 0, stream>>>(Wv, WvT, 2 * DMODEL, DMODEL);
  transpose_cast_kernel<<<4096, 256, 0, stream>>>(Wfc, WfcT, DMODEL, 2 * DMODEL);

  // projections (Q carries the 1/sqrt(dh)=0.125 scale — exact in bf16)
  gemm_bf16_kernel<<<dim3(32, 8), 256, 0, stream>>>(
      qbf, WqT, bq, Qp, (int)MTOK, DMODEL, DMODEL, 0, 0.125f);
  gemm_bf16_kernel<<<dim3(32, 8), 256, 0, stream>>>(
      kbf, WkT, bk, Kp, (int)MTOK, DMODEL, DMODEL, 0, 1.0f);
  gemm_bf16_kernel<<<dim3(32, 8), 256, 0, stream>>>(
      vbf, WvT, bv, Vt, (int)MTOK, DMODEL, 2 * DMODEL, 1, 1.0f);

  // attention + LDS-reduced head-averaged probabilities
  attn_kernel<<<dim3(S_LEN / 16, NBATCH), 512, 0, stream>>>(
      Qp, Kp, Vt, pmask, attn_avg, Obf);

  // final FC -> d_out (f32)
  gemm_bf16_kernel<<<dim3(32, 16), 256, 0, stream>>>(
      Obf, WfcT, bfc, outp, (int)MTOK, 2 * DMODEL, DMODEL, 3, 1.0f);
}
